// AttentionBlock_21388937134768
// MI455X (gfx1250) — matrix operand order repaired and checked by reading, not compiled, not run on hardware
//
#include <hip/hip_runtime.h>
#include <hip/hip_bf16.h>
#include <stdint.h>

typedef __attribute__((ext_vector_type(16))) _Float16 v16h;
typedef __attribute__((ext_vector_type(8)))  _Float16 v8h;
typedef __attribute__((ext_vector_type(8)))  float    v8f;

#define DIM   768
#define NSEQ  2048
#define NBAT  8
#define ROWS  (NBAT * NSEQ)   // 16384
#define KT    32              // K-chunk (WMMA K)
#define XPAD  40              // LDS row stride in halves (bank-conflict pad)

union HFrag { v16h v; v8h h[2]; };

// ---- monotonic float<->int key (for integer atomicMax on row maxima) ----
__device__ __forceinline__ int f2k(float f) {
    int i = __float_as_int(f);
    return i >= 0 ? i : (i ^ 0x7fffffff);
}
__device__ __forceinline__ float k2f(int k) {
    int i = k >= 0 ? k : (k ^ 0x7fffffff);
    return __int_as_float(i);
}

// =====================================================================
// Kernel 0: precision prep. x -> f16 row-major; W{q,k,v} -> f16 TRANSPOSED
// ([e][d]) so WMMA B-fragments are per-lane contiguous loads.
// =====================================================================
__global__ void prep_kernel(const float* __restrict__ x,
                            const float* __restrict__ Wq,
                            const float* __restrict__ Wk,
                            const float* __restrict__ Wv,
                            _Float16* __restrict__ xh,
                            _Float16* __restrict__ wt) {
    size_t tid    = (size_t)blockIdx.x * blockDim.x + threadIdx.x;
    size_t stride = (size_t)gridDim.x * blockDim.x;
    for (size_t i = tid; i < (size_t)ROWS * DIM; i += stride)
        xh[i] = (_Float16)x[i];
    for (size_t i = tid; i < (size_t)3 * DIM * DIM; i += stride) {
        int m = (int)(i / (DIM * DIM));
        int r = (int)((i % (DIM * DIM)) / DIM);  // e (out col)
        int c = (int)(i % DIM);                  // d (contraction)
        const float* W = (m == 0) ? Wq : (m == 1) ? Wk : Wv;
        wt[i] = (_Float16)W[(size_t)c * DIM + r];
    }
}

// =====================================================================
// Kernel 1: QKV projection, 128x128 output tile per workgroup, K double-
// buffered through LDS. grid = (ROWS/128, DIM/128, 3), block = 256 (8 waves).
// Wave w owns rows [w*16, w*16+16) x all 128 cols (8 accum tiles).
// z==0/1: store q/k row-major f16.  z==2: store V transposed [b][d][n] f16.
// =====================================================================
__global__ __launch_bounds__(256)
void proj_kernel(const _Float16* __restrict__ xh,
                 const _Float16* __restrict__ wt,
                 const float* __restrict__ bq, const float* __restrict__ bk,
                 const float* __restrict__ bv,
                 _Float16* __restrict__ qh, _Float16* __restrict__ kh,
                 _Float16* __restrict__ vt) {
    __shared__ _Float16 xs[2][128 * XPAD];   // x K-slice   (rows x KT)
    __shared__ _Float16 wl[2][128 * XPAD];   // W^T K-slice (cols x KT)

    const int tid  = threadIdx.x;
    const int lane = tid & 31;
    const int wv   = tid >> 5;
    const int grp  = lane >> 4;
    const int ln   = lane & 15;
    const int z    = blockIdx.z;
    const int r0   = blockIdx.x * 128;
    const int c0   = blockIdx.y * 128;

    const _Float16* W    = wt + (size_t)z * DIM * DIM;
    const float*    bias = (z == 0) ? bq : (z == 1) ? bk : bv;

    // staging map: thread -> (row 0..127, 16-half segment 0/16)
    const int srow = tid >> 1;
    const int sseg = (tid & 1) * 16;
    const _Float16* gx = xh + (size_t)(r0 + srow) * DIM + sseg;
    const _Float16* gw = W  + (size_t)(c0 + srow) * DIM + sseg;
    const int soff = srow * XPAD + sseg;

    v8f acc[8];
#pragma unroll
    for (int c = 0; c < 8; ++c) acc[c] = (v8f){};

    // preload chunk 0
    {
        v8h x0 = *(const v8h*)(gx);
        v8h x1 = *(const v8h*)(gx + 8);
        v8h w0 = *(const v8h*)(gw);
        v8h w1 = *(const v8h*)(gw + 8);
        *(v8h*)(&xs[0][soff])     = x0;
        *(v8h*)(&xs[0][soff + 8]) = x1;
        *(v8h*)(&wl[0][soff])     = w0;
        *(v8h*)(&wl[0][soff + 8]) = w1;
    }
    __syncthreads();

    int cur = 0;
    for (int dk = 0; dk < DIM / KT; ++dk) {
        // issue next chunk's global loads first (overlap with compute below)
        v8h x0, x1, w0, w1;
        const bool more = (dk + 1) < (DIM / KT);
        if (more) {
            const int d = (dk + 1) * KT;
            x0 = *(const v8h*)(gx + d);
            x1 = *(const v8h*)(gx + d + 8);
            w0 = *(const v8h*)(gw + d);
            w1 = *(const v8h*)(gw + d + 8);
        }

        // compute chunk dk from LDS buf[cur]
        HFrag a;
        const _Float16* xr = &xs[cur][(wv * 16 + ln) * XPAD];
        a.h[0] = *(const v8h*)(xr + 8 * grp);
        a.h[1] = *(const v8h*)(xr + 16 + 8 * grp);
#pragma unroll
        for (int c = 0; c < 8; ++c) {
            HFrag b;
            const _Float16* wr = &wl[cur][(c * 16 + ln) * XPAD];
            b.h[0] = *(const v8h*)(wr + 8 * grp);
            b.h[1] = *(const v8h*)(wr + 16 + 8 * grp);
            acc[c] = __builtin_amdgcn_wmma_f32_16x16x32_f16(false, a.v, false, b.v,
                                                            (short)0, acc[c], false, false);
        }

        if (more) {
            *(v8h*)(&xs[cur ^ 1][soff])     = x0;
            *(v8h*)(&xs[cur ^ 1][soff + 8]) = x1;
            *(v8h*)(&wl[cur ^ 1][soff])     = w0;
            *(v8h*)(&wl[cur ^ 1][soff + 8]) = w1;
        }
        __syncthreads();
        cur ^= 1;
    }

    if (z == 2) {
        // V transposed: vt[b][e][n]; lane writes 8 consecutive n per tile
        const int batch = r0 >> 11;                       // 2048 % 128 == 0
        const int nloc  = (r0 & 2047) + wv * 16 + 8 * grp;
#pragma unroll
        for (int c = 0; c < 8; ++c) {
            const float bval = bias[c0 + c * 16 + ln];
            v8h pv;
#pragma unroll
            for (int j = 0; j < 8; ++j) pv[j] = (_Float16)(acc[c][j] + bval);
            *(v8h*)(vt + (size_t)batch * DIM * NSEQ
                       + (size_t)(c0 + c * 16 + ln) * NSEQ + nloc) = pv;
        }
    } else {
        _Float16* dst = (z == 0) ? qh : kh;
#pragma unroll
        for (int c = 0; c < 8; ++c) {
            const float bval = bias[c0 + c * 16 + ln];
#pragma unroll
            for (int j = 0; j < 8; ++j) {
                const int row = r0 + wv * 16 + j + 8 * grp;
                dst[(size_t)row * DIM + c0 + c * 16 + ln] = (_Float16)(acc[c][j] + bval);
            }
        }
    }
}

// =====================================================================
// Kernel 2: fused flash attention, Br = 32 query rows per workgroup.
// grid = (NSEQ/32, NBAT), block = 256 (8 waves).
// Per iteration (128 keys): each wave scores a DIFFERENT 16-key tile against
// BOTH 16-row query sub-tiles (zero score redundancy); online-softmax stats
// merged via LDS atomics; P shared via LDS f16; then each wave accumulates
// P@V for one (query-half x 192-wide d-chunk) slab: 12 f32 accum tiles.
// =====================================================================
__global__ __launch_bounds__(256)
void attn_kernel(const _Float16* __restrict__ qh,
                 const _Float16* __restrict__ kh,
                 const _Float16* __restrict__ vt,
                 float* __restrict__ out) {
    __shared__ _Float16 P[8][32][16];   // [key tile][query row][key] 8 KB
    __shared__ int   m_run[32];
    __shared__ int   m_old[32];
    __shared__ float l_run[32];

    const int tid  = threadIdx.x;
    const int lane = tid & 31;
    const int wv   = tid >> 5;
    const int grp  = lane >> 4;
    const int ln   = lane & 15;
    const int qt   = blockIdx.x;   // 32-row query tile in batch
    const int b    = blockIdx.y;

    const int qhalf = wv >> 2;          // P@V: query half owned
    const int dch   = (wv & 3) * 192;   // P@V: d-chunk owned

    const _Float16* qrow0 = qh + ((size_t)b * NSEQ + qt * 32 + ln) * DIM;
    const _Float16* qrow1 = qrow0 + (size_t)16 * DIM;

    if (tid < 32) { m_run[tid] = f2k(-__builtin_inff()); l_run[tid] = 0.0f; }

    v8f o[12];
#pragma unroll
    for (int c = 0; c < 12; ++c) o[c] = (v8f){};

    for (int t = 0; t < NSEQ / 128; ++t) {
        __syncthreads();                         // prev P@V done; stats stable
        if (tid < 32) m_old[tid] = m_run[tid];
        __syncthreads();

        // ---- S = Q(32 rows) * K^T(16 keys) for this wave's key tile ----
        const int k0 = t * 128 + wv * 16;
        const _Float16* krow = kh + ((size_t)b * NSEQ + k0 + ln) * DIM;
        v8f s0 = {}, s1 = {};
        for (int dk = 0; dk < DIM / 32; ++dk) {
            const int d0 = dk * 32;
            HFrag a0, a1, bb;
            a0.h[0] = *(const v8h*)(qrow0 + d0 + 8 * grp);
            a0.h[1] = *(const v8h*)(qrow0 + d0 + 16 + 8 * grp);
            a1.h[0] = *(const v8h*)(qrow1 + d0 + 8 * grp);
            a1.h[1] = *(const v8h*)(qrow1 + d0 + 16 + 8 * grp);
            bb.h[0] = *(const v8h*)(krow + d0 + 8 * grp);
            bb.h[1] = *(const v8h*)(krow + d0 + 16 + 8 * grp);
            s0 = __builtin_amdgcn_wmma_f32_16x16x32_f16(false, a0.v, false, bb.v,
                                                        (short)0, s0, false, false);
            s1 = __builtin_amdgcn_wmma_f32_16x16x32_f16(false, a1.v, false, bb.v,
                                                        (short)0, s1, false, false);
        }
        const float sc = 0.03608439182435161f;   // 1/sqrt(768)
        float rm0[8], rm1[8];
#pragma unroll
        for (int j = 0; j < 8; ++j) {
            s0[j] *= sc; rm0[j] = s0[j];
            s1[j] *= sc; rm1[j] = s1[j];
        }
#pragma unroll
        for (int off = 8; off >= 1; off >>= 1)
#pragma unroll
            for (int j = 0; j < 8; ++j) {
                rm0[j] = fmaxf(rm0[j], __shfl_xor(rm0[j], off, 32));
                rm1[j] = fmaxf(rm1[j], __shfl_xor(rm1[j], off, 32));
            }
        if (ln == 0) {
#pragma unroll
            for (int j = 0; j < 8; ++j) {
                atomicMax(&m_run[j + 8 * grp],      f2k(rm0[j]));
                atomicMax(&m_run[16 + j + 8 * grp], f2k(rm1[j]));
            }
        }
        __syncthreads();
        if (tid < 32) {   // rescale old l before new sums arrive
            l_run[tid] *= __expf(k2f(m_old[tid]) - k2f(m_run[tid]));
        }
        __syncthreads();

        float mn0[8], mn1[8];
#pragma unroll
        for (int j = 0; j < 8; ++j) {
            mn0[j] = k2f(m_run[j + 8 * grp]);
            mn1[j] = k2f(m_run[16 + j + 8 * grp]);
        }
        float alpha[8];
#pragma unroll
        for (int j = 0; j < 8; ++j) {
            const int r = qhalf * 16 + j + 8 * grp;
            alpha[j] = __expf(k2f(m_old[r]) - k2f(m_run[r]));
        }
#pragma unroll
        for (int c = 0; c < 12; ++c)
#pragma unroll
            for (int j = 0; j < 8; ++j) o[c][j] *= alpha[j];

        float rs0[8], rs1[8];
#pragma unroll
        for (int j = 0; j < 8; ++j) {
            const float p0 = __expf(s0[j] - mn0[j]);
            const float p1 = __expf(s1[j] - mn1[j]);
            rs0[j] = p0; rs1[j] = p1;
            P[wv][j + 8 * grp][ln]      = (_Float16)p0;
            P[wv][16 + j + 8 * grp][ln] = (_Float16)p1;
        }
#pragma unroll
        for (int off = 8; off >= 1; off >>= 1)
#pragma unroll
            for (int j = 0; j < 8; ++j) {
                rs0[j] += __shfl_xor(rs0[j], off, 32);
                rs1[j] += __shfl_xor(rs1[j], off, 32);
            }
        if (ln == 0) {
#pragma unroll
            for (int j = 0; j < 8; ++j) {
                atomicAdd(&l_run[j + 8 * grp],      rs0[j]);
                atomicAdd(&l_run[16 + j + 8 * grp], rs1[j]);
            }
        }
        __syncthreads();

        // ---- O += P * V over this wave's (query-half, 192-wide d) slab ----
#pragma unroll
        for (int tp = 0; tp < 4; ++tp) {   // 2 P-tiles per WMMA (K = 32 keys)
            HFrag a;
            a.h[0] = *(const v8h*)(&P[2 * tp][qhalf * 16 + ln][8 * grp]);
            a.h[1] = *(const v8h*)(&P[2 * tp + 1][qhalf * 16 + ln][8 * grp]);
            const _Float16* vbase = vt + (size_t)b * DIM * NSEQ
                                  + (size_t)(t * 128 + tp * 32 + 8 * grp);
#pragma unroll
            for (int c = 0; c < 12; ++c) {
                const _Float16* vrow = vbase + (size_t)(dch + c * 16 + ln) * NSEQ;
                HFrag bb;
                bb.h[0] = *(const v8h*)(vrow);
                bb.h[1] = *(const v8h*)(vrow + 16);
                o[c] = __builtin_amdgcn_wmma_f32_16x16x32_f16(false, a.v, false, bb.v,
                                                              (short)0, o[c], false, false);
            }
        }
    }
    __syncthreads();

    // ---- normalize and write out (f32) ----
#pragma unroll
    for (int j = 0; j < 8; ++j) {
        const int   m   = qhalf * 16 + j + 8 * grp;
        const float inv = 1.0f / l_run[m];
        const size_t row = (size_t)b * NSEQ + qt * 32 + m;
#pragma unroll
        for (int c = 0; c < 12; ++c)
            out[row * DIM + dch + c * 16 + ln] = o[c][j] * inv;
    }
}

// =====================================================================
extern "C" void kernel_launch(void* const* d_in, const int* in_sizes, int n_in,
                              void* d_out, int out_size, void* d_ws, size_t ws_size,
                              hipStream_t stream) {
    (void)in_sizes; (void)n_in; (void)out_size; (void)ws_size;
    const float* x  = (const float*)d_in[0];
    const float* Wq = (const float*)d_in[1];
    const float* bq = (const float*)d_in[2];
    const float* Wk = (const float*)d_in[3];
    const float* bk = (const float*)d_in[4];
    const float* Wv = (const float*)d_in[5];
    const float* bv = (const float*)d_in[6];
    float* out = (float*)d_out;

    _Float16* xh = (_Float16*)d_ws;                 // [ROWS][DIM]
    _Float16* qh = xh + (size_t)ROWS * DIM;         // [ROWS][DIM]
    _Float16* kh = qh + (size_t)ROWS * DIM;         // [ROWS][DIM]
    _Float16* vt = kh + (size_t)ROWS * DIM;         // [NBAT][DIM][NSEQ]
    _Float16* wt = vt + (size_t)ROWS * DIM;         // [3][DIM][DIM] transposed

    prep_kernel<<<2048, 256, 0, stream>>>(x, Wq, Wk, Wv, xh, wt);

    dim3 g1(ROWS / 128, DIM / 128, 3);
    proj_kernel<<<g1, 256, 0, stream>>>(xh, wt, bq, bk, bv, qh, kh, vt);

    dim3 g2(NSEQ / 32, NBAT);
    attn_kernel<<<g2, 256, 0, stream>>>(qh, kh, vt, out);
}
